// VectorQuantizerEMA_1460288881297
// MI455X (gfx1250) — compile-verified, matrix operand order checked by reading, not run on hardware
//
#include <hip/hip_runtime.h>
#include <hip/hip_bf16.h>

// ---------------------------------------------------------------------------
// VQ-VAE EMA vector quantizer for MI455X (gfx1250, wave32).
// Core: f16 WMMA (v_wmma_f32_16x16x32_f16) streaming argmax over K=8192
// codewords; 256KB z slab DMA'd into LDS via GLOBAL_LOAD_ASYNC_TO_LDS_B128
// (ASYNCcnt-tracked), cb rows streamed from L2 through the vmem pipe.
// ---------------------------------------------------------------------------

typedef __attribute__((ext_vector_type(4)))  _Float16 v4h;
typedef __attribute__((ext_vector_type(8)))  _Float16 v8h;
typedef __attribute__((ext_vector_type(16))) _Float16 v16h;
typedef __attribute__((ext_vector_type(8)))  float    v8f;

constexpr int    NB = 32768;   // batch rows
constexpr int    NK = 8192;    // codewords
constexpr int    ND = 1024;    // feature dim
constexpr int    ROWS_PER_BLOCK = 128;   // 8 waves x 16 rows

// Assemble a v16h operand from the two 16-byte chunks demanded by the
// gfx1250 16-bit A/B operand layout (dwords 0-3 = K+0..7, dwords 4-7 = K+16..23
// for the low lane-half; +8 shift for the high half handled by caller).
__device__ __forceinline__ v16h ld16(const _Float16* p) {
  union { struct { v8h lo, hi; } s; v16h v; } u;
  u.s.lo = *(const v8h*)(p);
  u.s.hi = *(const v8h*)(p + 16);
  return u.v;
}

// ---------------------------------------------------------------------------
// Kernel 1: row L2-normalize f32 -> f16 (one wave per row).
// ---------------------------------------------------------------------------
__global__ void __launch_bounds__(256)
rownorm_f16_kernel(const float* __restrict__ in, _Float16* __restrict__ out,
                   int nrows) {
  const int lane = threadIdx.x & 31;
  const int row  = blockIdx.x * (blockDim.x >> 5) + (threadIdx.x >> 5);
  if (row >= nrows) return;

  const float4* r = (const float4*)(in + (size_t)row * ND);
  float4 v[8];
  float ss = 0.f;
#pragma unroll
  for (int i = 0; i < 8; ++i) {
    v[i] = r[lane + i * 32];
    ss += v[i].x * v[i].x + v[i].y * v[i].y + v[i].z * v[i].z + v[i].w * v[i].w;
  }
#pragma unroll
  for (int m = 16; m >= 1; m >>= 1) ss += __shfl_xor(ss, m, 32);
  const float s = 1.0f / fmaxf(sqrtf(ss), 1e-12f);

  v4h* o = (v4h*)(out + (size_t)row * ND);
#pragma unroll
  for (int i = 0; i < 8; ++i) {
    v4h h;
    h[0] = (_Float16)(v[i].x * s);
    h[1] = (_Float16)(v[i].y * s);
    h[2] = (_Float16)(v[i].z * s);
    h[3] = (_Float16)(v[i].w * s);
    o[lane + i * 32] = h;
  }
}

// ---------------------------------------------------------------------------
// Kernel 2: WMMA streaming argmax.  Block = 8 waves, 128 z rows.
// z slab (128 x 1024 f16 = 256 KB) is DMA'd into LDS with async global->LDS
// transfers (no VGPR round-trip); each wave owns 16 rows and sweeps all K
// codewords, 4 N-tiles (64 codewords) per pass so the A operand (ds_load) is
// reused 4x per WMMA group while B streams through the vmem pipe.
// ---------------------------------------------------------------------------
__global__ void __launch_bounds__(256)
vq_argmax_kernel(const _Float16* __restrict__ znorm,
                 const _Float16* __restrict__ cbnorm,
                 int* __restrict__ code_ids) {
  extern __shared__ _Float16 smem[];   // ROWS_PER_BLOCK * ND halves

  const int tid  = threadIdx.x;
  const int lane = tid & 31;
  const int wave = tid >> 5;
  const size_t row0 = (size_t)blockIdx.x * ROWS_PER_BLOCK;

  // ---- async DMA stage: 256 KB global -> LDS, ASYNCcnt-tracked ----
  // Each thread issues 64 x b128 async transfers (hardware throttles the
  // 6-bit ASYNCcnt).  LDS operand is the 32-bit LDS byte offset;
  // global operand is a 64-bit VGPR pair (GV mode).
  {
    const uint32_t lds_base = (uint32_t)(uintptr_t)smem;
    const uint64_t g_base   = (uint64_t)(uintptr_t)(znorm + row0 * ND);
    const int n16 = (ROWS_PER_BLOCK * ND * 2) / 16;   // 16384 x 16B
    for (int i = tid; i < n16; i += 256) {
      const uint32_t ldsa = lds_base + (uint32_t)i * 16u;
      const uint64_t ga   = g_base + (uint64_t)i * 16u;
      asm volatile("global_load_async_to_lds_b128 %0, %1, off"
                   :: "v"(ldsa), "v"(ga) : "memory");
    }
    asm volatile("s_wait_asynccnt 0" ::: "memory");
  }
  __syncthreads();

  const int half  = lane >> 4;        // which 16-lane half
  const int hoff  = half * 8;         // K-offset shift for high half
  const int nlane = lane & 15;        // A row / B column within tile

  const _Float16* aRow  = smem   + (size_t)(wave * 16 + nlane) * ND + hoff;
  const _Float16* bBase = cbnorm + (size_t)nlane * ND + hoff;

  float bestVal[8];
  int   bestIdx[8];
#pragma unroll
  for (int j = 0; j < 8; ++j) { bestVal[j] = -3.4e38f; bestIdx[j] = 0; }

  for (int nt = 0; nt < NK; nt += 64) {
    const _Float16* b0 = bBase + (size_t)nt * ND;
    const _Float16* b1 = b0 + (size_t)16 * ND;
    const _Float16* b2 = b0 + (size_t)32 * ND;
    const _Float16* b3 = b0 + (size_t)48 * ND;
    // hint the next codeword group toward L1/L2 (global_prefetch_b8)
    __builtin_prefetch(b0 + (size_t)64 * ND, 0, 1);

    v8f c0 = {}, c1 = {}, c2 = {}, c3 = {};
    for (int d = 0; d < ND; d += 32) {
      const v16h a = ld16(aRow + d);
      c0 = __builtin_amdgcn_wmma_f32_16x16x32_f16(false, a, false, ld16(b0 + d),
                                                  (short)0, c0, false, false);
      c1 = __builtin_amdgcn_wmma_f32_16x16x32_f16(false, a, false, ld16(b1 + d),
                                                  (short)0, c1, false, false);
      c2 = __builtin_amdgcn_wmma_f32_16x16x32_f16(false, a, false, ld16(b2 + d),
                                                  (short)0, c2, false, false);
      c3 = __builtin_amdgcn_wmma_f32_16x16x32_f16(false, a, false, ld16(b3 + d),
                                                  (short)0, c3, false, false);
    }

    // running argmax update; C layout: vgpr j <-> row j (+8 for high half),
    // column = nt_tile + (lane & 15).  Ascending scan + strict '>' keeps the
    // first (lowest-index) maximum like jnp.argmax.
    const int n = nt + nlane;
#pragma unroll
    for (int j = 0; j < 8; ++j) {
      if (c0[j] > bestVal[j]) { bestVal[j] = c0[j]; bestIdx[j] = n;      }
      if (c1[j] > bestVal[j]) { bestVal[j] = c1[j]; bestIdx[j] = n + 16; }
      if (c2[j] > bestVal[j]) { bestVal[j] = c2[j]; bestIdx[j] = n + 32; }
      if (c3[j] > bestVal[j]) { bestVal[j] = c3[j]; bestIdx[j] = n + 48; }
    }
  }

  // reduce over the 16 lanes of each half (xor masks <= 8 keep halves apart)
#pragma unroll
  for (int j = 0; j < 8; ++j) {
    float v = bestVal[j];
    int idx = bestIdx[j];
#pragma unroll
    for (int m = 8; m >= 1; m >>= 1) {
      const float ov = __shfl_xor(v, m, 32);
      const int   oi = __shfl_xor(idx, m, 32);
      if (ov > v || (ov == v && oi < idx)) { v = ov; idx = oi; }
    }
    if (nlane == 0)
      code_ids[row0 + (size_t)(wave * 16 + half * 8 + j)] = idx;
  }
}

// ---------------------------------------------------------------------------
// Kernel 3: zero stats region (n_k, loss accum, sum_k).
// ---------------------------------------------------------------------------
__global__ void zero_kernel(float* __restrict__ p, size_t n) {
  const size_t i = (size_t)blockIdx.x * blockDim.x + threadIdx.x;
  if (i < n) p[i] = 0.f;
}

// ---------------------------------------------------------------------------
// Kernel 4: gather q, write z_q (= q under STE), commit-loss partial sums,
// segment-sum scatter via f32 global atomics (one wave per row).
// ---------------------------------------------------------------------------
__global__ void __launch_bounds__(256)
gather_scatter_kernel(const float* __restrict__ z,
                      const float* __restrict__ codebook,
                      const int* __restrict__ ids,
                      float* __restrict__ zq,
                      float* __restrict__ ids_f,
                      float* __restrict__ nk,
                      float* __restrict__ sumk,
                      float* __restrict__ loss_acc) {
  const int lane = threadIdx.x & 31;
  const int row  = blockIdx.x * (blockDim.x >> 5) + (threadIdx.x >> 5);
  if (row >= NB) return;

  const int id = ids[row];
  if (lane == 0) {
    ids_f[row] = (float)id;
    atomicAdd(&nk[id], 1.0f);
  }

  const float* zr = z        + (size_t)row * ND;
  const float* cr = codebook + (size_t)id  * ND;
  float*       qr = zq       + (size_t)row * ND;
  float*       sr = sumk     + (size_t)id  * ND;

  float lsum = 0.f;
  for (int i = lane; i < ND; i += 32) {
    const float zv = zr[i];
    const float qv = cr[i];
    qr[i] = qv;                       // z + sg(q - z) == q numerically
    const float dd = zv - qv;
    lsum += dd * dd;
    atomicAdd(&sr[i], zv);
  }
#pragma unroll
  for (int m = 16; m >= 1; m >>= 1) lsum += __shfl_xor(lsum, m, 32);
  if (lane == 0) atomicAdd(loss_acc, lsum);
}

// ---------------------------------------------------------------------------
// Kernel 5: EMA finalize + loss scale.
// ---------------------------------------------------------------------------
__global__ void __launch_bounds__(256)
finalize_kernel(const float* __restrict__ ema_count,
                const float* __restrict__ ema_weight,
                const float* __restrict__ nk,
                const float* __restrict__ sumk,
                const float* __restrict__ loss_acc,
                float* __restrict__ out_codebook,
                float* __restrict__ out_count,
                float* __restrict__ out_weight,
                float* __restrict__ out_loss) {
  const size_t idx = (size_t)blockIdx.x * blockDim.x + threadIdx.x;
  if (idx >= (size_t)NK * ND) return;
  const size_t k = idx >> 10;   // ND == 1024
  const float nc = 0.99f * ema_count[k] + 0.01f * nk[k];
  const float nw = 0.99f * ema_weight[idx] + 0.01f * sumk[idx];
  out_weight[idx]   = nw;
  out_codebook[idx] = nw / (nc + 1e-5f);
  if ((idx & 1023) == 0) out_count[k] = nc;
  if (idx == 0) out_loss[0] = 0.25f * loss_acc[0] / ((float)NB * (float)ND);
}

// ---------------------------------------------------------------------------
extern "C" void kernel_launch(void* const* d_in, const int* in_sizes, int n_in,
                              void* d_out, int out_size, void* d_ws, size_t ws_size,
                              hipStream_t stream) {
  const float* z          = (const float*)d_in[0];
  const float* codebook   = (const float*)d_in[1];
  const float* ema_count  = (const float*)d_in[2];
  const float* ema_weight = (const float*)d_in[3];

  // workspace layout
  char* ws = (char*)d_ws;
  _Float16* znorm  = (_Float16*)ws;                                  // NB*ND f16
  _Float16* cbnorm = (_Float16*)(ws + (size_t)NB * ND * 2);          // NK*ND f16
  int*      ids    = (int*)(ws + (size_t)NB * ND * 2 + (size_t)NK * ND * 2);
  float*    stats  = (float*)((char*)ids + (size_t)NB * 4);
  float*    nk       = stats;                                        // NK
  float*    loss_acc = stats + NK;                                   // 1 (+pad)
  float*    sumk     = stats + NK + 64;                              // NK*ND
  const size_t statsN = (size_t)NK + 64 + (size_t)NK * ND;

  // output layout (reference return order, flat f32)
  float* o            = (float*)d_out;
  float* out_zq       = o;                                           // NB*ND
  float* out_ids      = o + (size_t)NB * ND;                         // NB
  float* out_loss     = out_ids + NB;                                // 1
  float* out_codebook = out_loss + 1;                                // NK*ND
  float* out_count    = out_codebook + (size_t)NK * ND;              // NK
  float* out_weight   = out_count + NK;                              // NK*ND

  rownorm_f16_kernel<<<NB / 8, 256, 0, stream>>>(z, znorm, NB);
  rownorm_f16_kernel<<<NK / 8, 256, 0, stream>>>(codebook, cbnorm, NK);
  zero_kernel<<<(int)((statsN + 255) / 256), 256, 0, stream>>>(stats, statsN);

  const size_t ldsBytes = (size_t)ROWS_PER_BLOCK * ND * sizeof(_Float16); // 256 KB
  vq_argmax_kernel<<<NB / ROWS_PER_BLOCK, 256, ldsBytes, stream>>>(znorm, cbnorm, ids);

  gather_scatter_kernel<<<NB / 8, 256, 0, stream>>>(z, codebook, ids, out_zq,
                                                    out_ids, nk, sumk, loss_acc);
  finalize_kernel<<<(NK * ND) / 256, 256, 0, stream>>>(ema_count, ema_weight,
                                                       nk, sumk, loss_acc,
                                                       out_codebook, out_count,
                                                       out_weight, out_loss);
}
